// Generator_71502615544318
// MI455X (gfx1250) — matrix operand order repaired and checked by reading, not compiled, not run on hardware
//
#include <hip/hip_runtime.h>
#include <hip/hip_bf16.h>

// ---------------- types & helpers ----------------
typedef __attribute__((ext_vector_type(16))) __bf16 v16bf;
typedef __attribute__((ext_vector_type(8)))  float  v8f;
typedef unsigned int u32x4 __attribute__((ext_vector_type(4)));

union Frag16 {
    v16bf v;
    unsigned short u[16];
    u32x4 q[2];
};
union FragF { float f[8]; v8f v; };

__device__ __forceinline__ unsigned short f2bf(float f) {
    unsigned int x = __float_as_uint(f);
    unsigned int r = (x + 0x7FFFu + ((x >> 16) & 1u)) >> 16;
    return (unsigned short)r;
}
__device__ __forceinline__ float bf2f(unsigned short u) {
    return __uint_as_float(((unsigned int)u) << 16);
}
__device__ __forceinline__ u32x4 ld4(const unsigned short* p) {
    return *reinterpret_cast<const u32x4*>(p);
}
__device__ __forceinline__ v8f wmma_bf16(const Frag16& a, const Frag16& b, v8f c) {
    return __builtin_amdgcn_wmma_f32_16x16x32_bf16(false, a.v, false, b.v,
                                                   (short)0, c, false, false);
}

// scale grid: L = 1024 (32x32) + 784 (28x28) + 625 (25x25) = 2433
#define LTOT 2433
#define LPAD 2464   // LTOT padded to multiple of 32 (and 16B-friendly leading dim)
__device__ __forceinline__ void loc_of(int l, int& off, int& hs, int& ws) {
    if (l < 1024)      { off = 0;    hs = 32; ws = 32; }
    else if (l < 1808) { off = 1024; hs = 28; ws = 28; }
    else               { off = 1808; hs = 25; ws = 25; }
}

// ---------------- converts ----------------
__global__ void k_f2bf(const float* __restrict__ in, unsigned short* __restrict__ out, int n) {
    int i = blockIdx.x * blockDim.x + threadIdx.x;
    if (i < n) out[i] = f2bf(in[i]);
}

// [B][512][1024] f32 -> position-major [B][1024][512] bf16
__global__ void k_transpose_pm(const float* __restrict__ in, unsigned short* __restrict__ out) {
    int idx = blockIdx.x * blockDim.x + threadIdx.x;
    if (idx >= 4 * 1024 * 512) return;
    int c = idx & 511;
    int pos = (idx >> 9) & 1023;
    int b = idx >> 19;
    out[idx] = f2bf(in[((size_t)b * 512 + c) * 1024 + pos]);
}

// ---------------- bicubic resize (Keys a=-0.5, half-pixel, edge renormalized) ----------------
__device__ __forceinline__ float cubw(float x) {
    x = fabsf(x);
    const float a = -0.5f;
    if (x <= 1.f) return ((a + 2.f) * x - (a + 3.f)) * x * x + 1.f;
    if (x <  2.f) return (((x - 5.f) * x + 8.f) * x - 4.f) * a;
    return 0.f;
}

// output position-major: out[b][pos][c], pos over hs*ws
__global__ void k_resize(const float* __restrict__ in, unsigned short* __restrict__ out,
                         int hs, int ws) {
    int idx = blockIdx.x * blockDim.x + threadIdx.x;
    int hw = hs * ws;
    int total = 4 * 512 * hw;
    if (idx >= total) return;
    int c = idx & 511;
    int t = idx >> 9;
    int pos = t % hw;
    int b = t / hw;
    int y = pos / ws, x = pos % ws;
    float sy = (y + 0.5f) * (32.f / (float)hs) - 0.5f;
    float sx = (x + 0.5f) * (32.f / (float)ws) - 0.5f;
    int iy = (int)floorf(sy), ix = (int)floorf(sx);
    float fy = sy - (float)iy, fx = sx - (float)ix;
    const float* src = in + ((size_t)b * 512 + c) * 1024;
    float acc = 0.f, wsum = 0.f;
    for (int j = -1; j <= 2; ++j) {
        int yy = iy + j; float wy = cubw(fy - (float)j);
        for (int i = -1; i <= 2; ++i) {
            int xx = ix + i; float wx = cubw(fx - (float)i);
            if (yy >= 0 && yy < 32 && xx >= 0 && xx < 32) {
                float w = wy * wx;
                acc += w * src[yy * 32 + xx];
                wsum += w;
            }
        }
    }
    out[idx] = f2bf(wsum != 0.f ? acc / wsum : 0.f);
}

// ---------------- 1x1-conv GEMM: Ypm = prelu(W[O,512] * Xpm^T + bias), all b128 ----------------
__global__ void __launch_bounds__(32)
k_conv1x1_wmma(const unsigned short* __restrict__ Wb,   // [O][512] bf16 row-major
               const unsigned short* __restrict__ Xpm,  // [N][512] bf16 position-major
               const float* __restrict__ bias,          // [O]
               const float* __restrict__ alpha,         // [1]
               int O, int N, int ldO, int ypos0,
               unsigned short* __restrict__ Ypm)        // [ypos0+N rows][ldO] bf16
{
    const int lane = threadIdx.x & 31;
    const int half = lane >> 4, lr = lane & 15;
    const int n0 = blockIdx.x * 16, o0 = blockIdx.y * 16;
    const int o = o0 + lr;
    const int n = n0 + lr;
    const int nq = (n < N) ? n : (N - 1);          // clamp; garbage cols never stored
    const unsigned short* aRow = Wb + (size_t)o * 512;
    const unsigned short* bRow = Xpm + (size_t)nq * 512;
    v8f acc = {};
    for (int k0 = 0; k0 < 512; k0 += 32) {
        Frag16 af, bfm;
        int ka = k0 + half * 8;
        af.q[0] = ld4(aRow + ka);
        af.q[1] = ld4(aRow + ka + 16);
        int kb = k0 + half * 8;
        bfm.q[0] = ld4(bRow + kb);
        bfm.q[1] = ld4(bRow + kb + 16);
        acc = wmma_bf16(af, bfm, acc);
    }
    if (n < N) {
        const float al = alpha[0];
        FragF d; d.v = acc;
        unsigned short* yrow = Ypm + (size_t)(ypos0 + n) * ldO + o0 + half * 8;
#pragma unroll
        for (int r = 0; r < 8; ++r) {
            float v = d.f[r] + bias[o0 + half * 8 + r];
            v = (v >= 0.f) ? v : al * v;
            yrow[r] = f2bf(v);
        }
    }
}

// ---------------- per-(b,l) patch L2 norm -> 1/max(norm, 1e-4) ----------------
__global__ void k_norm(const unsigned short* __restrict__ m_pm,   // [B][LTOT][256]
                       float* __restrict__ inv_norm)              // [B][LPAD]
{
    __shared__ float red[256];
    int l = blockIdx.x, b = blockIdx.y, c = threadIdx.x;
    int off, hs, ws; loc_of(l, off, hs, ws);
    int p = l - off, yl = p / ws, xl = p % ws;
    const unsigned short* mb = m_pm + (size_t)b * LTOT * 256;
    float s = 0.f;
    for (int t = 0; t < 9; ++t) {
        int y2 = yl + t / 3 - 1, x2 = xl + t % 3 - 1;
        if (y2 >= 0 && y2 < hs && x2 >= 0 && x2 < ws) {
            float v = bf2f(mb[(size_t)(off + y2 * ws + x2) * 256 + c]);
            s += v * v;
        }
    }
    red[c] = s; __syncthreads();
    for (int st = 128; st > 0; st >>= 1) {
        if (c < st) red[c] += red[c + st];
        __syncthreads();
    }
    if (c == 0) {
        float nr = fmaxf(sqrtf(red[0]), 1e-4f);
        inv_norm[b * LPAD + l] = 1.f / nr;
    }
}

// ---------------- attention scores, 2x2 register-tiled WMMA ----------------
// each wave: 32 l-rows x 32 pixels; K = 9 taps x 256 ch; 2 b128 loads per wmma
__global__ void __launch_bounds__(32)
k_scores_wmma(const unsigned short* __restrict__ m_pm,    // [B][LTOT][256]
              const unsigned short* __restrict__ mb_pm,   // [B][1024][256]
              const float* __restrict__ inv_norm,         // [B][LPAD]
              float* __restrict__ sbuf)                   // [B][1024][LPAD] (n-major)
{
    const int lane = threadIdx.x & 31;
    const int half = lane >> 4, lr = lane & 15;
    const int b = blockIdx.z;
    const int n0 = blockIdx.x * 32, l0 = blockIdx.y * 32;
    // two pixel columns and two l rows per lane
    const int nA0 = n0 + lr,      nA1 = n0 + 16 + lr;
    const int qy0 = nA0 >> 5, qx0 = nA0 & 31;
    const int qy1 = nA1 >> 5, qx1 = nA1 & 31;
    const int lA0 = l0 + lr,      lA1 = l0 + 16 + lr;
    const int lc0 = (lA0 < LTOT) ? lA0 : (LTOT - 1);
    const int lc1 = (lA1 < LTOT) ? lA1 : (LTOT - 1);
    int of0, hs0, ws0; loc_of(lc0, of0, hs0, ws0);
    int of1, hs1, ws1; loc_of(lc1, of1, hs1, ws1);
    const int p0 = lc0 - of0, yl0 = p0 / ws0, xl0 = p0 % ws0;
    const int p1 = lc1 - of1, yl1 = p1 / ws1, xl1 = p1 % ws1;
    const bool lv0 = (lA0 < LTOT), lv1 = (lA1 < LTOT);
    const unsigned short* mb = m_pm  + (size_t)b * LTOT * 256;
    const unsigned short* qb = mb_pm + (size_t)b * 1024 * 256;
    const u32x4 z4 = {0u, 0u, 0u, 0u};
    v8f acc00 = {}, acc01 = {}, acc10 = {}, acc11 = {};
    for (int t = 0; t < 9; ++t) {
        const int di = t / 3, dj = t % 3;
        // A (key patches) rows
        int ya0 = yl0 + di - 1, xa0 = xl0 + dj - 1;
        bool av0 = lv0 && ya0 >= 0 && ya0 < hs0 && xa0 >= 0 && xa0 < ws0;
        const unsigned short* aRow0 = mb + (size_t)(av0 ? (of0 + ya0 * ws0 + xa0) : 0) * 256;
        int ya1 = yl1 + di - 1, xa1 = xl1 + dj - 1;
        bool av1 = lv1 && ya1 >= 0 && ya1 < hs1 && xa1 >= 0 && xa1 < ws1;
        const unsigned short* aRow1 = mb + (size_t)(av1 ? (of1 + ya1 * ws1 + xa1) : 0) * 256;
        // B (query patches) columns
        int by0 = qy0 + di - 1, bx0 = qx0 + dj - 1;
        bool bv0 = by0 >= 0 && by0 < 32 && bx0 >= 0 && bx0 < 32;
        const unsigned short* bRow0 = qb + (size_t)(bv0 ? (by0 * 32 + bx0) : 0) * 256;
        int by1 = qy1 + di - 1, bx1 = qx1 + dj - 1;
        bool bv1 = by1 >= 0 && by1 < 32 && bx1 >= 0 && bx1 < 32;
        const unsigned short* bRow1 = qb + (size_t)(bv1 ? (by1 * 32 + bx1) : 0) * 256;
        for (int cb = 0; cb < 256; cb += 32) {
            const int ca = cb + half * 8;
            const int ck = cb + half * 8;
            Frag16 a0, a1, b0, b1;
            a0.q[0] = av0 ? ld4(aRow0 + ca)      : z4;
            a0.q[1] = av0 ? ld4(aRow0 + ca + 16) : z4;
            a1.q[0] = av1 ? ld4(aRow1 + ca)      : z4;
            a1.q[1] = av1 ? ld4(aRow1 + ca + 16) : z4;
            b0.q[0] = bv0 ? ld4(bRow0 + ck)      : z4;
            b0.q[1] = bv0 ? ld4(bRow0 + ck + 16) : z4;
            b1.q[0] = bv1 ? ld4(bRow1 + ck)      : z4;
            b1.q[1] = bv1 ? ld4(bRow1 + ck + 16) : z4;
            acc00 = wmma_bf16(a0, b0, acc00);
            acc01 = wmma_bf16(a0, b1, acc01);
            acc10 = wmma_bf16(a1, b0, acc10);
            acc11 = wmma_bf16(a1, b1, acc11);
        }
    }
    // stores: 8 consecutive l per lane; pad rows [LTOT, LPAD) never read
    const v8f accs[4] = {acc00, acc01, acc10, acc11};
#pragma unroll
    for (int i = 0; i < 2; ++i) {          // l tile
#pragma unroll
        for (int j = 0; j < 2; ++j) {      // n tile
            FragF d; d.v = accs[i * 2 + j];
            const int nn = n0 + 16 * j + lr;
            const int lbase = l0 + 16 * i + half * 8;
            float* srow = sbuf + ((size_t)b * 1024 + nn) * LPAD + lbase;
            const float* inr = inv_norm + b * LPAD + lbase;
#pragma unroll
            for (int r = 0; r < 8; ++r) srow[r] = d.f[r] * inr[r];
        }
    }
}

// ---------------- softmax over L per (b, pixel); writes bf16 + zeroed pad ----------------
__global__ void k_softmax(const float* __restrict__ sbuf,        // [B][1024][LPAD]
                          unsigned short* __restrict__ abuf)     // [B][1024][LPAD] bf16
{
    __shared__ float red[256];
    int n = blockIdx.x, b = blockIdx.y, tid = threadIdx.x;
    const float* s = sbuf + ((size_t)b * 1024 + n) * LPAD;
    float m = -1e30f;
    for (int l = tid; l < LTOT; l += 256) m = fmaxf(m, s[l]);
    red[tid] = m; __syncthreads();
    for (int st = 128; st > 0; st >>= 1) {
        if (tid < st) red[tid] = fmaxf(red[tid], red[tid + st]);
        __syncthreads();
    }
    m = red[0]; __syncthreads();
    float sum = 0.f;
    for (int l = tid; l < LTOT; l += 256) sum += expf(10.f * (s[l] - m));
    red[tid] = sum; __syncthreads();
    for (int st = 128; st > 0; st >>= 1) {
        if (tid < st) red[tid] += red[tid + st];
        __syncthreads();
    }
    float inv = 1.f / red[0];
    unsigned short* a = abuf + ((size_t)b * 1024 + n) * LPAD;
    for (int l = tid; l < LPAD; l += 256) {
        unsigned short v = 0;
        if (l < LTOT) v = f2bf(expf(10.f * (s[l] - m)) * inv);
        a[l] = v;
    }
}

// ---------------- expand flipped-kernel raw patches: pbase[b][t][o][lpad] ----------------
__global__ void k_expand_base(const unsigned short* __restrict__ base_pm,  // [B][LTOT][512]
                              unsigned short* __restrict__ pbase)          // [B][9][512][LPAD]
{
    int idx = blockIdx.x * blockDim.x + threadIdx.x;
    if (idx >= 4 * 9 * 512 * LPAD) return;
    int lp = idx % LPAD;
    int t2 = idx / LPAD;
    int o  = t2 & 511;
    int t3 = t2 >> 9;
    int t  = t3 % 9;
    int b  = t3 / 9;
    unsigned short v = 0;
    if (lp < LTOT) {
        int off, hs, ws; loc_of(lp, off, hs, ws);
        int p = lp - off, yl = p / ws, xl = p % ws;
        int y2 = yl + t / 3 - 1, x2 = xl + t % 3 - 1;
        if (y2 >= 0 && y2 < hs && x2 >= 0 && x2 < ws)
            v = base_pm[((size_t)b * LTOT + off + y2 * ws + x2) * 512 + o];
    }
    pbase[idx] = v;
}

// ---------------- assembly, 2x2 register-tiled WMMA ----------------
// each wave: 32 pixels x 32 channels; K = 9 * LPAD; 2 b128 loads per wmma
__global__ void __launch_bounds__(32)
k_assemble_wmma(const unsigned short* __restrict__ pbase,   // [B][9][512][LPAD]
                const unsigned short* __restrict__ abuf,    // [B][1024][LPAD] bf16
                const float* __restrict__ input,            // [B][512][1024]
                float* __restrict__ out)                    // [B][512][1024]
{
    const int lane = threadIdx.x & 31;
    const int half = lane >> 4, lr = lane & 15;
    const int b = blockIdx.z;
    const int n0 = blockIdx.x * 32, o0 = blockIdx.y * 32;
    const int nA0 = n0 + lr,      nA1 = n0 + 16 + lr;       // A rows: pixels
    const int qy0 = nA0 >> 5, qx0 = nA0 & 31;
    const int qy1 = nA1 >> 5, qx1 = nA1 & 31;
    const int oB0 = o0 + lr,      oB1 = o0 + 16 + lr;       // B cols: channels
    const unsigned short* ab = abuf + (size_t)b * 1024 * LPAD;
    const u32x4 z4 = {0u, 0u, 0u, 0u};
    v8f acc00 = {}, acc01 = {}, acc10 = {}, acc11 = {};
    for (int t = 0; t < 9; ++t) {
        const int di = t / 3, dj = t % 3;
        // conv_transpose == conv with flipped kernel: attention sampled at (1-di, 1-dj)
        int ay0 = qy0 + 1 - di, ax0 = qx0 + 1 - dj;
        bool av0 = ay0 >= 0 && ay0 < 32 && ax0 >= 0 && ax0 < 32;
        const unsigned short* aRow0 = ab + (size_t)(av0 ? (ay0 * 32 + ax0) : 0) * LPAD;
        int ay1 = qy1 + 1 - di, ax1 = qx1 + 1 - dj;
        bool av1 = ay1 >= 0 && ay1 < 32 && ax1 >= 0 && ax1 < 32;
        const unsigned short* aRow1 = ab + (size_t)(av1 ? (ay1 * 32 + ax1) : 0) * LPAD;
        const unsigned short* pbt = pbase + ((size_t)b * 9 + t) * 512 * LPAD;
        const unsigned short* bRow0 = pbt + (size_t)oB0 * LPAD;
        const unsigned short* bRow1 = pbt + (size_t)oB1 * LPAD;
        for (int lb = 0; lb < LPAD; lb += 32) {
            const int la = lb + half * 8;
            const int lk = lb + half * 8;
            Frag16 a0, a1, b0, b1;
            a0.q[0] = av0 ? ld4(aRow0 + la)      : z4;
            a0.q[1] = av0 ? ld4(aRow0 + la + 16) : z4;
            a1.q[0] = av1 ? ld4(aRow1 + la)      : z4;
            a1.q[1] = av1 ? ld4(aRow1 + la + 16) : z4;
            b0.q[0] = ld4(bRow0 + lk);
            b0.q[1] = ld4(bRow0 + lk + 16);
            b1.q[0] = ld4(bRow1 + lk);
            b1.q[1] = ld4(bRow1 + lk + 16);
            acc00 = wmma_bf16(a0, b0, acc00);
            acc01 = wmma_bf16(a0, b1, acc01);
            acc10 = wmma_bf16(a1, b0, acc10);
            acc11 = wmma_bf16(a1, b1, acc11);
        }
    }
    const v8f accs[4] = {acc00, acc01, acc10, acc11};
#pragma unroll
    for (int i = 0; i < 2; ++i) {          // pixel tile (D rows)
#pragma unroll
        for (int j = 0; j < 2; ++j) {      // channel tile (D cols)
            FragF d; d.v = accs[i * 2 + j];
            const int oo = o0 + 16 * j + lr;
            size_t base = ((size_t)b * 512 + oo) * 1024 + n0 + 16 * i + half * 8;
#pragma unroll
            for (int r = 0; r < 8; ++r) out[base + r] = d.f[r] * 0.25f + input[base + r];
        }
    }
}

// ---------------- host ----------------
extern "C" void kernel_launch(void* const* d_in, const int* in_sizes, int n_in,
                              void* d_out, int out_size, void* d_ws, size_t ws_size,
                              hipStream_t stream)
{
    (void)in_sizes; (void)n_in; (void)out_size; (void)ws_size;
    const float* input   = (const float*)d_in[0];
    const float* w_base  = (const float*)d_in[1];
    const float* b_base  = (const float*)d_in[2];
    const float* a_base  = (const float*)d_in[3];
    const float* w_match = (const float*)d_in[4];
    const float* b_match = (const float*)d_in[5];
    const float* a_match = (const float*)d_in[6];
    const float* w_asm   = (const float*)d_in[7];
    const float* b_asm   = (const float*)d_in[8];
    const float* a_asm   = (const float*)d_in[9];
    float* out = (float*)d_out;

    char* ws = (char*)d_ws;
    size_t ofs = 0;
    auto alloc = [&](size_t bytes) -> void* {
        void* p = ws + ofs;
        ofs = (ofs + bytes + 255) & ~(size_t)255;
        return p;
    };
    unsigned short* in_pm    = (unsigned short*)alloc((size_t)4 * 1024 * 512 * 2);
    unsigned short* ref1_pm  = (unsigned short*)alloc((size_t)4 * 784 * 512 * 2);
    unsigned short* ref2_pm  = (unsigned short*)alloc((size_t)4 * 625 * 512 * 2);
    unsigned short* wb_base  = (unsigned short*)alloc((size_t)256 * 512 * 2);
    unsigned short* wb_match = (unsigned short*)alloc((size_t)256 * 512 * 2);
    unsigned short* wb_asm   = (unsigned short*)alloc((size_t)512 * 512 * 2);
    unsigned short* mb_pm    = (unsigned short*)alloc((size_t)4 * 1024 * 256 * 2);
    unsigned short* m_pm     = (unsigned short*)alloc((size_t)4 * LTOT * 256 * 2);
    unsigned short* base_pm  = (unsigned short*)alloc((size_t)4 * LTOT * 512 * 2);
    float*          inv_norm = (float*)alloc((size_t)4 * LPAD * 4);
    float*          sbuf     = (float*)alloc((size_t)4 * 1024 * LPAD * 4);
    unsigned short* abuf     = (unsigned short*)alloc((size_t)4 * 1024 * LPAD * 2);
    unsigned short* pbase    = (unsigned short*)alloc((size_t)4 * 9 * 512 * LPAD * 2);

    // converts / pyramid
    { int n = 4 * 1024 * 512; k_transpose_pm<<<(n + 255) / 256, 256, 0, stream>>>(input, in_pm); }
    { int n = 256 * 512;      k_f2bf<<<(n + 255) / 256, 256, 0, stream>>>(w_base, wb_base, n); }
    { int n = 256 * 512;      k_f2bf<<<(n + 255) / 256, 256, 0, stream>>>(w_match, wb_match, n); }
    { int n = 512 * 512;      k_f2bf<<<(n + 255) / 256, 256, 0, stream>>>(w_asm, wb_asm, n); }
    { int n = 4 * 512 * 784;  k_resize<<<(n + 255) / 256, 256, 0, stream>>>(input, ref1_pm, 28, 28); }
    { int n = 4 * 512 * 625;  k_resize<<<(n + 255) / 256, 256, 0, stream>>>(input, ref2_pm, 25, 25); }

    const int Ns[3]   = {1024, 784, 625};
    const int offs[3] = {0, 1024, 1808};
    for (int b = 0; b < 4; ++b) {
        const unsigned short* refs[3] = {
            in_pm   + (size_t)b * 1024 * 512,
            ref1_pm + (size_t)b * 784 * 512,
            ref2_pm + (size_t)b * 625 * 512
        };
        // match_base = prelu(w_base * input) -> [1024][256]
        k_conv1x1_wmma<<<dim3(64, 16, 1), 32, 0, stream>>>(
            wb_base, refs[0], b_base, a_base, 256, 1024, 256, 0,
            mb_pm + (size_t)b * 1024 * 256);
        for (int sc = 0; sc < 3; ++sc) {
            int N = Ns[sc], gx = (N + 15) / 16;
            k_conv1x1_wmma<<<dim3(gx, 16, 1), 32, 0, stream>>>(
                wb_match, refs[sc], b_match, a_match, 256, N, 256, offs[sc],
                m_pm + (size_t)b * LTOT * 256);
            k_conv1x1_wmma<<<dim3(gx, 32, 1), 32, 0, stream>>>(
                wb_asm, refs[sc], b_asm, a_asm, 512, N, 512, offs[sc],
                base_pm + (size_t)b * LTOT * 512);
        }
    }
    k_norm<<<dim3(LTOT, 4, 1), 256, 0, stream>>>(m_pm, inv_norm);
    k_scores_wmma<<<dim3(32, 77, 4), 32, 0, stream>>>(m_pm, mb_pm, inv_norm, sbuf);
    k_softmax<<<dim3(1024, 4, 1), 256, 0, stream>>>(sbuf, abuf);
    { int n = 4 * 9 * 512 * LPAD; k_expand_base<<<(n + 255) / 256, 256, 0, stream>>>(base_pm, pbase); }
    k_assemble_wmma<<<dim3(32, 16, 4), 32, 0, stream>>>(pbase, abuf, input, out);
}
